// NeighborhoodAttentionModule_6923487282395
// MI455X (gfx1250) — compile-verified, matrix operand order checked by reading, not run on hardware
//
#include <hip/hip_runtime.h>
#include <math.h>

// ---- CDNA5 WMMA fragment types (wave32) ----
typedef __attribute__((ext_vector_type(16))) __bf16 bf16x16;
typedef __attribute__((ext_vector_type(8)))  float   floatx8;

#define D_EMB 256   // embed dim
#define D_QK  128   // total q/k dim (2 heads x 64)
#define KNB   16    // neighbors per row
#define TB    16    // batch rows per block

// Element counts of the bf16-packed weights in the workspace
#define WQ_ELEMS (D_QK * D_EMB)          // 32768
#define WK_ELEMS (D_QK * D_EMB)          // 32768
#define WG_ELEMS (D_EMB * 2 * D_EMB)     // 131072

// Build a 16x32 bf16 A-fragment from f32 memory (global or LDS).
// ISA 7.12.2 layout: lanes 0-15 hold row=lane, K = {0..7, 16..23};
// lanes 16-31 hold row=lane-16, K = {8..15, 24..31}.
__device__ __forceinline__ bf16x16 frag_a_f32(const float* __restrict__ base,
                                              int row_stride, int k0, int lane) {
    const int row  = lane & 15;
    const int koff = (lane & 16) ? 8 : 0;
    const float* p = base + row * row_stride + k0 + koff;
    float4 q0 = *reinterpret_cast<const float4*>(p);
    float4 q1 = *reinterpret_cast<const float4*>(p + 4);
    float4 q2 = *reinterpret_cast<const float4*>(p + 16);
    float4 q3 = *reinterpret_cast<const float4*>(p + 20);
    bf16x16 v;
    v[0]  = (__bf16)q0.x; v[1]  = (__bf16)q0.y; v[2]  = (__bf16)q0.z; v[3]  = (__bf16)q0.w;
    v[4]  = (__bf16)q1.x; v[5]  = (__bf16)q1.y; v[6]  = (__bf16)q1.z; v[7]  = (__bf16)q1.w;
    v[8]  = (__bf16)q2.x; v[9]  = (__bf16)q2.y; v[10] = (__bf16)q2.z; v[11] = (__bf16)q2.w;
    v[12] = (__bf16)q3.x; v[13] = (__bf16)q3.y; v[14] = (__bf16)q3.z; v[15] = (__bf16)q3.w;
    return v;
}

// Load a pre-packed bf16 B-fragment: layout [nt][kc][lane][16 bf16],
// i.e. each lane's 16 elements are 32 contiguous bytes -> 2x global_load_b128,
// and a wave's fragment is one contiguous, fully-coalesced 1KB burst.
__device__ __forceinline__ bf16x16 frag_b_pk(const unsigned short* __restrict__ w,
                                             int KC, int nt, int kc, int lane) {
    const uint4* p = reinterpret_cast<const uint4*>(
        w + ((size_t)(((nt * KC) + kc) * 32 + lane) << 4));
    union { bf16x16 v; uint4 q[2]; } r;
    r.q[0] = p[0];
    r.q[1] = p[1];
    return r.v;
}

__device__ __forceinline__ floatx8 wmma_bf16(bf16x16 a, bf16x16 b, floatx8 c) {
    // (neg_a, A, neg_b, B, c_mod, C, reuse_a, reuse_b)
    return __builtin_amdgcn_wmma_f32_16x16x32_bf16(false, a, false, b, (short)0, c,
                                                   false, false);
}

// ---- one-shot weight repack: f32 [Nout][Kin] -> bf16 fragment order ----------
__global__ void pack_weights_bf16(const float* __restrict__ W,
                                  unsigned short* __restrict__ dst,
                                  int Nout, int Kin) {
    const int KC    = Kin / 32;
    const int total = (Nout / 16) * KC * 32;   // lane-slots
    int idx = blockIdx.x * blockDim.x + threadIdx.x;
    if (idx >= total) return;
    const int lane = idx & 31;
    const int kc   = (idx >> 5) % KC;
    const int nt   = (idx >> 5) / KC;
    const int n     = nt * 16 + (lane & 15);
    const int kbase = kc * 32 + ((lane & 16) ? 8 : 0);
    const float* src = W + (size_t)n * Kin + kbase;
    union { __bf16 b[16]; uint4 q[2]; } r;
    #pragma unroll
    for (int e = 0; e < 8; ++e) r.b[e]     = (__bf16)src[e];
    #pragma unroll
    for (int e = 0; e < 8; ++e) r.b[8 + e] = (__bf16)src[16 + e];
    uint4* d = reinterpret_cast<uint4*>(dst) + (size_t)idx * 2;
    d[0] = r.q[0];
    d[1] = r.q[1];
}

__global__ __launch_bounds__(256, 1)
void na_fused_kernel(const float* __restrict__ center,
                     const float* __restrict__ neighbor,
                     const float* __restrict__ wts,
                     const unsigned short* __restrict__ wq,   // packed bf16
                     const unsigned short* __restrict__ wk,   // packed bf16
                     const unsigned short* __restrict__ wg,   // packed bf16
                     const float* __restrict__ bg,
                     float* __restrict__ out) {
    const int tid  = threadIdx.x;
    const int lane = tid & 31;
    const int wv   = tid >> 5;          // wave id 0..7
    const int col  = lane & 15;         // column within 16-wide tile
    const int hi16 = (lane >> 4) & 1;   // 0: rows 0..7, 1: rows 8..15 of C tiles
    const int b0   = blockIdx.x * TB;

    __shared__ __align__(16) float sC[TB][D_EMB];    // staged center tile
    __shared__ __align__(16) float sAgg[TB][D_EMB];  // aggregated neighbors
    __shared__ __align__(16) float sQ[TB][D_QK];     // q projection (pre-scaled 1/8)
    __shared__ float sScore[TB][2][KNB];
    __shared__ float sAttn[TB][2][KNB];
    __shared__ float sWc[TB][KNB];                   // head-mean attention weights

    // ================= Phase 0: stage center, zero scores, Q projection =========
    {
        const float4* gsrc = reinterpret_cast<const float4*>(center + (size_t)b0 * D_EMB);
        float4*       ldst = reinterpret_cast<float4*>(&sC[0][0]);
        for (int i = tid; i < TB * D_EMB / 4; i += 256) ldst[i] = gsrc[i];
        float* sf = &sScore[0][0][0];
        for (int i = tid; i < TB * 2 * KNB; i += 256) sf[i] = 0.0f;

        // each wave computes one 16x16 tile of Q = center @ Wq^T
        floatx8 acc = (floatx8)0.0f;
        #pragma unroll
        for (int kc = 0; kc < D_EMB / 32; ++kc) {
            bf16x16 a = frag_a_f32(center + (size_t)b0 * D_EMB, D_EMB, kc * 32, lane);
            bf16x16 b = frag_b_pk(wq, D_EMB / 32, wv, kc, lane);
            acc = wmma_bf16(a, b, acc);
        }
        const int n0 = wv * 16;
        #pragma unroll
        for (int g = 0; g < 8; ++g) {
            int M = g + hi16 * 8;
            sQ[M][n0 + col] = acc[g] * 0.125f;   // 1/sqrt(64)
        }
    }
    __syncthreads();

    // ================= Phase 1: K projection (dominant GEMM) + scores ===========
    // wave w handles batch rows mt = 2w, 2w+1; M-tile = that row's 16 neighbors.
    for (int mt = 2 * wv; mt < 2 * wv + 2; ++mt) {
        const float* nb = neighbor + ((size_t)(b0 + mt) * KNB) * D_EMB;
        floatx8 acc[8];
        #pragma unroll
        for (int nt = 0; nt < 8; ++nt) acc[nt] = (floatx8)0.0f;

        for (int kc = 0; kc < D_EMB / 32; ++kc) {
            // prefetch next k-chunk of the A tile (global_prefetch_b8)
            if (kc + 1 < D_EMB / 32)
                __builtin_prefetch(nb + (lane & 15) * D_EMB + (kc + 1) * 32, 0, 0);
            bf16x16 a = frag_a_f32(nb, D_EMB, kc * 32, lane);
            #pragma unroll
            for (int nt = 0; nt < 8; ++nt) {
                bf16x16 b = frag_b_pk(wk, D_EMB / 32, nt, kc, lane);
                acc[nt] = wmma_bf16(a, b, acc[nt]);
            }
        }
        // scores[b,h,kn] += sum_a k[kn, h*64+a] * q[b, h*64+a]
        #pragma unroll
        for (int nt = 0; nt < 8; ++nt) {
            const int n0 = nt * 16;
            const int h  = n0 >> 6;               // tile fully inside one head
            const float qv = sQ[mt][n0 + col];
            #pragma unroll
            for (int g = 0; g < 8; ++g) {
                float v = acc[nt][g] * qv;
                v += __shfl_xor(v, 1, 32);
                v += __shfl_xor(v, 2, 32);
                v += __shfl_xor(v, 4, 32);
                v += __shfl_xor(v, 8, 32);        // sum over 16 columns
                if (col == 0) {
                    int kn = g + hi16 * 8;
                    atomicAdd(&sScore[mt][h][kn], v);   // ds_add_f32
                }
            }
        }
    }
    __syncthreads();

    // ================= Phase 2: softmax with log-prior and zero-mask ============
    if (tid < TB * 2) {
        int bl = tid >> 1, h = tid & 1;
        float sc[KNB];
        float mx = -INFINITY;
        for (int kn = 0; kn < KNB; ++kn) {
            float w = wts[(size_t)(b0 + bl) * KNB + kn];
            float s = sScore[bl][h][kn];
            s = (w > 0.0f) ? s + __logf(fmaxf(w, 1e-8f)) : -INFINITY;
            sc[kn] = s;
            mx = fmaxf(mx, s);
        }
        float sum = 0.0f;
        for (int kn = 0; kn < KNB; ++kn) {
            float e = (sc[kn] == -INFINITY) ? 0.0f : __expf(sc[kn] - mx);
            sc[kn] = e;
            sum += e;
        }
        float inv = (sum > 0.0f) ? 1.0f / sum : 0.0f;
        for (int kn = 0; kn < KNB; ++kn) sAttn[bl][h][kn] = sc[kn] * inv;
    }
    __syncthreads();
    if (tid < TB) {
        for (int kn = 0; kn < KNB; ++kn)
            sWc[tid][kn] = 0.5f * (sAttn[tid][0][kn] + sAttn[tid][1][kn]);
    }
    __syncthreads();

    // ================= Phase 3: aggregate raw neighbors (float4 VALU) ===========
    {
        int bl = tid >> 4, c16 = tid & 15;
        const float4* nb4 =
            reinterpret_cast<const float4*>(neighbor + ((size_t)(b0 + bl) * KNB) * D_EMB);
        float4* agg4 = reinterpret_cast<float4*>(&sAgg[bl][0]);
        #pragma unroll
        for (int i = 0; i < 4; ++i) {
            int d4 = c16 + 16 * i;
            float4 a = {0.f, 0.f, 0.f, 0.f};
            for (int kn = 0; kn < KNB; ++kn) {
                float w = sWc[bl][kn];
                float4 v = nb4[kn * (D_EMB / 4) + d4];
                a.x += w * v.x; a.y += w * v.y; a.z += w * v.z; a.w += w * v.w;
            }
            agg4[d4] = a;
        }
    }
    __syncthreads();

    // ================= Phase 4: gate GEMM (Kdim=512) + sigmoid blend ============
    {
        floatx8 acc[2];
        acc[0] = (floatx8)0.0f;
        acc[1] = (floatx8)0.0f;
        for (int kc = 0; kc < (2 * D_EMB) / 32; ++kc) {
            int k0 = kc * 32;
            bf16x16 a = (k0 < D_EMB)
                            ? frag_a_f32(center + (size_t)b0 * D_EMB, D_EMB, k0, lane)
                            : frag_a_f32(&sAgg[0][0], D_EMB, k0 - D_EMB, lane);
            #pragma unroll
            for (int tt = 0; tt < 2; ++tt) {
                bf16x16 b = frag_b_pk(wg, (2 * D_EMB) / 32, 2 * wv + tt, kc, lane);
                acc[tt] = wmma_bf16(a, b, acc[tt]);
            }
        }
        #pragma unroll
        for (int tt = 0; tt < 2; ++tt) {
            int n0 = (2 * wv + tt) * 16;
            int n  = n0 + col;
            float bias = bg[n];
            #pragma unroll
            for (int g = 0; g < 8; ++g) {
                int M = g + hi16 * 8;
                float x    = acc[tt][g] + bias;
                float gate = 1.0f / (1.0f + __expf(-x));
                float c    = sC[M][n];
                float a    = sAgg[M][n];
                out[(size_t)(b0 + M) * D_EMB + n] = gate * c + (1.0f - gate) * a;
            }
        }
    }
}

extern "C" void kernel_launch(void* const* d_in, const int* in_sizes, int n_in,
                              void* d_out, int out_size, void* d_ws, size_t ws_size,
                              hipStream_t stream) {
    const float* center   = (const float*)d_in[0];
    const float* neighbor = (const float*)d_in[1];
    const float* wts      = (const float*)d_in[2];
    const float* Wq       = (const float*)d_in[3];
    const float* Wk       = (const float*)d_in[4];
    const float* Wg       = (const float*)d_in[5];
    const float* bg       = (const float*)d_in[6];
    float* out            = (float*)d_out;

    unsigned short* wsq = (unsigned short*)d_ws;
    unsigned short* wsk = wsq + WQ_ELEMS;
    unsigned short* wsg = wsk + WK_ELEMS;

    // repack weights to bf16 fragment order (runs every call: deterministic)
    {
        const int slots_qk = (D_QK / 16) * (D_EMB / 32) * 32;        // 2048
        const int slots_g  = (D_EMB / 16) * ((2 * D_EMB) / 32) * 32; // 8192
        pack_weights_bf16<<<(slots_qk + 255) / 256, 256, 0, stream>>>(Wq, wsq, D_QK, D_EMB);
        pack_weights_bf16<<<(slots_qk + 255) / 256, 256, 0, stream>>>(Wk, wsk, D_QK, D_EMB);
        pack_weights_bf16<<<(slots_g + 255) / 256, 256, 0, stream>>>(Wg, wsg, D_EMB, 2 * D_EMB);
    }

    const int Bn   = in_sizes[0] / D_EMB;   // 32768
    const int grid = Bn / TB;               // 2048 blocks of 16 rows
    na_fused_kernel<<<grid, 256, 0, stream>>>(center, neighbor, wts, wsq, wsk, wsg, bg, out);
}
